// Attention_28338194219036
// MI455X (gfx1250) — compile-verified, matrix operand order checked by reading, not run on hardware
//
#include <hip/hip_runtime.h>
#include <hip/hip_bf16.h>
#include <math.h>

#define N_HEADS_    16
#define N_KV_HEADS_ 4
#define HEAD_DIM_   128

typedef __attribute__((ext_vector_type(16))) __bf16 bf16x16;
typedef __attribute__((ext_vector_type(8)))  float  f32x8;

union FragU { bf16x16 v; uint4 q[2]; };

// Load a 16x32 bf16 WMMA fragment from a row-major matrix (ld in elements).
// A-matrix: "row" indexes M; B-matrix: "row" indexes N (lane n holds column n of B).
// ISA layout: lanes 0-15 hold K=0..7 (+16..23), lanes 16-31 hold K=8..15 (+24..31).
__device__ __forceinline__ bf16x16 load_frag(const __bf16* __restrict__ base,
                                             int ld, int row0, int k0, int lane) {
  const int l  = lane & 15;
  const int hi = lane >> 4;
  const __bf16* p = base + (size_t)(row0 + l) * ld + k0 + hi * 8;
  FragU f;
  f.q[0] = *(const uint4*)(p);
  f.q[1] = *(const uint4*)(p + 16);
  return f.v;
}

__device__ __forceinline__ f32x8 wmma_bf16(bf16x16 a, bf16x16 b, f32x8 c) {
  return __builtin_amdgcn_wmma_f32_16x16x32_bf16(false, a, false, b, (short)0, c,
                                                 false, false);
}

// ---------------------------------------------------------------------------
// fp32 -> bf16 bulk convert
// ---------------------------------------------------------------------------
__global__ void cvt_f32_bf16(const float* __restrict__ in, __bf16* __restrict__ out,
                             size_t n) {
  size_t i = (size_t)blockIdx.x * blockDim.x + threadIdx.x;
  const size_t stride = (size_t)gridDim.x * blockDim.x;
  for (; i < n; i += stride) out[i] = (__bf16)in[i];
}

// ---------------------------------------------------------------------------
// C[M,N] (f32) = A[M,K] (bf16, row-major) * W[N,K]^T (bf16, row-major)
// Block: 256 threads = 8 waves in 4x2, block tile 128x128, wave tile 32x64.
// Manually double-buffered K-loop: next iteration's 6 fragment loads are in
// flight while the current 8 WMMAs execute (partial s_wait_loadcnt overlap).
// ---------------------------------------------------------------------------
__global__ __launch_bounds__(256) void gemm_bf16(
    const __bf16* __restrict__ A, const __bf16* __restrict__ W,
    float* __restrict__ C, int M, int N, int Kd)
{
  const int lane = threadIdx.x & 31;
  const int wave = threadIdx.x >> 5;
  const int m0 = blockIdx.y * 128 + (wave >> 1) * 32;  // 4 waves down M
  const int n0 = blockIdx.x * 128 + (wave & 1) * 64;   // 2 waves across N
  const int l = lane & 15, hi = lane >> 4;

  f32x8 acc[2][4] = {};

  // prologue: fragments for k0 = 0
  bf16x16 a0 = load_frag(A, Kd, m0,      0, lane);
  bf16x16 a1 = load_frag(A, Kd, m0 + 16, 0, lane);
  bf16x16 b0 = load_frag(W, Kd, n0,      0, lane);
  bf16x16 b1 = load_frag(W, Kd, n0 + 16, 0, lane);
  bf16x16 b2 = load_frag(W, Kd, n0 + 32, 0, lane);
  bf16x16 b3 = load_frag(W, Kd, n0 + 48, 0, lane);

  for (int k0 = 0; k0 < Kd; k0 += 32) {
    // issue next-iteration loads first (wraps to 0 on last iter: redundant,
    // branch-free, keeps the pipeline full)
    const int kn = (k0 + 32 < Kd) ? k0 + 32 : 0;
    bf16x16 na0 = load_frag(A, Kd, m0,      kn, lane);
    bf16x16 na1 = load_frag(A, Kd, m0 + 16, kn, lane);
    bf16x16 nb0 = load_frag(W, Kd, n0,      kn, lane);
    bf16x16 nb1 = load_frag(W, Kd, n0 + 16, kn, lane);
    bf16x16 nb2 = load_frag(W, Kd, n0 + 32, kn, lane);
    bf16x16 nb3 = load_frag(W, Kd, n0 + 48, kn, lane);
    __builtin_prefetch(A + (size_t)(m0 + l) * Kd + k0 + 128, 0, 3);
    __builtin_prefetch(W + (size_t)(n0 + l) * Kd + k0 + 128, 0, 3);

    acc[0][0] = wmma_bf16(a0, b0, acc[0][0]);
    acc[1][0] = wmma_bf16(a1, b0, acc[1][0]);
    acc[0][1] = wmma_bf16(a0, b1, acc[0][1]);
    acc[1][1] = wmma_bf16(a1, b1, acc[1][1]);
    acc[0][2] = wmma_bf16(a0, b2, acc[0][2]);
    acc[1][2] = wmma_bf16(a1, b2, acc[1][2]);
    acc[0][3] = wmma_bf16(a0, b3, acc[0][3]);
    acc[1][3] = wmma_bf16(a1, b3, acc[1][3]);

    a0 = na0; a1 = na1; b0 = nb0; b1 = nb1; b2 = nb2; b3 = nb3;
  }
#pragma unroll
  for (int i = 0; i < 2; ++i)
#pragma unroll
    for (int j = 0; j < 4; ++j)
#pragma unroll
      for (int r = 0; r < 8; ++r)
        C[(size_t)(m0 + i * 16 + r + 8 * hi) * N + n0 + j * 16 + l] = acc[i][j][r];
}

// ---------------------------------------------------------------------------
// RMSNorm + RoPE on q,k; transpose-convert v.
// grid = (T, 24, B): slot 0..15 = Q head, 16..19 = K head, 20..23 = V head.
// block = 128 threads (one per head-dim element).
// ---------------------------------------------------------------------------
__global__ __launch_bounds__(128) void prep_kernel(
    const float* __restrict__ qf, const float* __restrict__ kf,
    const float* __restrict__ vf,
    __bf16* __restrict__ Q, __bf16* __restrict__ K, __bf16* __restrict__ Vt, int T)
{
  const int t = blockIdx.x;
  const int slot = blockIdx.y;
  const int b = blockIdx.z;
  const int d = threadIdx.x;
  const size_t row = (size_t)b * T + t;

  if (slot >= 20) {  // V: just convert + transpose to [b, kh, d, T]
    const int kh = slot - 20;
    const float v = vf[row * (N_KV_HEADS_ * HEAD_DIM_) + kh * HEAD_DIM_ + d];
    Vt[((size_t)(b * N_KV_HEADS_ + kh) * HEAD_DIM_ + d) * T + t] = (__bf16)v;
    return;
  }

  __shared__ float red[HEAD_DIM_];
  __shared__ float snorm[HEAD_DIM_];
  const bool isQ = slot < 16;
  const int hh = isQ ? slot : slot - 16;
  const float* src = isQ ? (qf + row * (N_HEADS_ * HEAD_DIM_) + hh * HEAD_DIM_)
                         : (kf + row * (N_KV_HEADS_ * HEAD_DIM_) + hh * HEAD_DIM_);
  const float v = src[d];
  red[d] = v * v;
  __syncthreads();
#pragma unroll
  for (int sft = 64; sft > 0; sft >>= 1) {
    if (d < sft) red[d] += red[d + sft];
    __syncthreads();
  }
  const float inv_rms = rsqrtf(red[0] * (1.0f / HEAD_DIM_) + 1.1920929e-7f);
  snorm[d] = v * inv_rms;
  __syncthreads();

  const int i = d & 63;
  // inv_freq = 10000^(-2i/128)
  const float invf = __expf(-(float)(2 * i) * (1.0f / HEAD_DIM_) * 9.2103403719761836f);
  const float ang = (float)t * invf;
  const float c = __cosf(ang);
  const float s = __sinf(ang);
  const float x1 = snorm[i];
  const float x2 = snorm[i + 64];
  const float outv = (d < 64) ? (x1 * c + x2 * s) : (x2 * c - x1 * s);

  __bf16* dst = isQ ? (Q + ((size_t)(b * N_HEADS_ + hh) * T + t) * HEAD_DIM_)
                    : (K + ((size_t)(b * N_KV_HEADS_ + hh) * T + t) * HEAD_DIM_);
  dst[d] = (__bf16)outv;
}

// ---------------------------------------------------------------------------
// Flash-attention: grid = (T/128, H, B), block = 128 threads = 4 waves.
// Each wave owns TWO 16-row query bands (32 rows): every K/V fragment fetched
// from memory is reused by 2 WMMAs (32 B/lane per WMMA). Online softmax with
// cross-lane shfl reductions; P relayout C->A via per-wave LDS tile. Softmax
// VALU co-executes with XDL WMMAs (16-bit WMMA tracked as TRANS on CDNA5).
// ---------------------------------------------------------------------------
__global__ __launch_bounds__(128) void attn_kernel(
    const __bf16* __restrict__ Q, const __bf16* __restrict__ K,
    const __bf16* __restrict__ Vt, __bf16* __restrict__ Y, int T)
{
  __shared__ __bf16 lds_p[4][32 * 32];
  const int lane = threadIdx.x & 31;
  const int wave = threadIdx.x >> 5;
  const int l = lane & 15;
  const int hi = lane >> 4;
  const int b = blockIdx.z;
  const int h = blockIdx.y;
  const int kh = h >> 2;  // 16 heads / 4 kv heads
  const int q0 = blockIdx.x * 128 + wave * 32;

  const __bf16* Qbh = Q + ((size_t)(b * N_HEADS_ + h) * T) * HEAD_DIM_;
  const __bf16* Kbh = K + ((size_t)(b * N_KV_HEADS_ + kh) * T) * HEAD_DIM_;
  const __bf16* Vbh = Vt + ((size_t)(b * N_KV_HEADS_ + kh) * HEAD_DIM_) * T;

  bf16x16 qf[2][4];
#pragma unroll
  for (int bnd = 0; bnd < 2; ++bnd)
#pragma unroll
    for (int kk = 0; kk < 4; ++kk)
      qf[bnd][kk] = load_frag(Qbh, HEAD_DIM_, q0 + bnd * 16, kk * 32, lane);

  f32x8 yacc[2][8] = {};
  float mrow[2][8], lrow[2][8];
#pragma unroll
  for (int bnd = 0; bnd < 2; ++bnd)
#pragma unroll
    for (int r = 0; r < 8; ++r) { mrow[bnd][r] = -1e30f; lrow[bnd][r] = 0.0f; }

  const float scale = 0.088388347648318447f;  // 1/sqrt(128)

  for (int kc = 0; kc < q0 + 32; kc += 32) {
    // L2 prefetch for the next key chunk
    __builtin_prefetch(Kbh + (size_t)(kc + 32 + l) * HEAD_DIM_, 0, 3);
    __builtin_prefetch(Vbh + (size_t)(l * 16) * T + kc + 32, 0, 3);

    // --- S = Q Kt for this 32-key chunk: 8 K-frag loads feed 16 WMMAs ---
    f32x8 s[2][2] = {};
#pragma unroll
    for (int kk = 0; kk < 4; ++kk) {
      bf16x16 k0f = load_frag(Kbh, HEAD_DIM_, kc,      kk * 32, lane);
      bf16x16 k1f = load_frag(Kbh, HEAD_DIM_, kc + 16, kk * 32, lane);
#pragma unroll
      for (int bnd = 0; bnd < 2; ++bnd) {
        s[bnd][0] = wmma_bf16(qf[bnd][kk], k0f, s[bnd][0]);
        s[bnd][1] = wmma_bf16(qf[bnd][kk], k1f, s[bnd][1]);
      }
    }
    // --- causal mask + online softmax (rows live in 16-lane N groups) ---
    float corr[2][8];
#pragma unroll
    for (int bnd = 0; bnd < 2; ++bnd) {
#pragma unroll
      for (int r = 0; r < 8; ++r) {
        const int qrow = q0 + bnd * 16 + r + 8 * hi;
        float v0 = (kc + l      <= qrow) ? s[bnd][0][r] * scale : -1e30f;
        float v1 = (kc + 16 + l <= qrow) ? s[bnd][1][r] * scale : -1e30f;
        float mx = fmaxf(v0, v1);
#pragma unroll
        for (int off = 8; off > 0; off >>= 1) mx = fmaxf(mx, __shfl_xor(mx, off, 32));
        const float mnew = fmaxf(mrow[bnd][r], mx);
        corr[bnd][r] = __expf(mrow[bnd][r] - mnew);
        const float e0 = __expf(v0 - mnew);
        const float e1 = __expf(v1 - mnew);
        float ps = e0 + e1;
#pragma unroll
        for (int off = 8; off > 0; off >>= 1) ps += __shfl_xor(ps, off, 32);
        lrow[bnd][r] = lrow[bnd][r] * corr[bnd][r] + ps;
        mrow[bnd][r] = mnew;
        // spill P (C layout) to LDS row-major for A-fragment reload
        lds_p[wave][(bnd * 16 + r + 8 * hi) * 32 + l]      = (__bf16)e0;
        lds_p[wave][(bnd * 16 + r + 8 * hi) * 32 + 16 + l] = (__bf16)e1;
      }
    }
#pragma unroll
    for (int bnd = 0; bnd < 2; ++bnd)
#pragma unroll
      for (int j = 0; j < 8; ++j)
#pragma unroll
        for (int r = 0; r < 8; ++r) yacc[bnd][j][r] *= corr[bnd][r];

    // --- y += P V: 8 V-frag loads feed 16 WMMAs (Vt rows contiguous in T) ---
    bf16x16 pf0 = load_frag(&lds_p[wave][0], 32, 0,  0, lane);
    bf16x16 pf1 = load_frag(&lds_p[wave][0], 32, 16, 0, lane);
#pragma unroll
    for (int j = 0; j < 8; ++j) {
      bf16x16 vfrag = load_frag(Vbh, T, j * 16, kc, lane);
      yacc[0][j] = wmma_bf16(pf0, vfrag, yacc[0][j]);
      yacc[1][j] = wmma_bf16(pf1, vfrag, yacc[1][j]);
    }
  }

  // --- normalize and store to Y[b, t, h*128 + d] as bf16 ---
#pragma unroll
  for (int bnd = 0; bnd < 2; ++bnd) {
#pragma unroll
    for (int j = 0; j < 8; ++j) {
#pragma unroll
      for (int r = 0; r < 8; ++r) {
        const float val = yacc[bnd][j][r] / lrow[bnd][r];
        Y[((size_t)b * T + q0 + bnd * 16 + r + 8 * hi) * (N_HEADS_ * HEAD_DIM_) +
          h * HEAD_DIM_ + j * 16 + l] = (__bf16)val;
      }
    }
  }
}

// ---------------------------------------------------------------------------
extern "C" void kernel_launch(void* const* d_in, const int* in_sizes, int n_in,
                              void* d_out, int out_size, void* d_ws, size_t ws_size,
                              hipStream_t stream)
{
  (void)in_sizes; (void)n_in; (void)out_size; (void)ws_size;
  const int B = 2, T = 2048, Cdim = 2048;
  const int KV = N_KV_HEADS_ * HEAD_DIM_;  // 512
  const size_t M = (size_t)B * T;          // 4096

  const float* x  = (const float*)d_in[0];
  const float* Wq = (const float*)d_in[1];
  const float* Wk = (const float*)d_in[2];
  const float* Wv = (const float*)d_in[3];
  const float* Wo = (const float*)d_in[4];

  char* wp = (char*)d_ws;
  auto alloc = [&](size_t bytes) {
    void* p = (void*)wp;
    wp += (bytes + 255) & ~(size_t)255;
    return p;
  };

  __bf16* xb  = (__bf16*)alloc(M * Cdim * 2);
  __bf16* wqb = (__bf16*)alloc((size_t)Cdim * Cdim * 2);
  __bf16* wkb = (__bf16*)alloc((size_t)KV * Cdim * 2);
  __bf16* wvb = (__bf16*)alloc((size_t)KV * Cdim * 2);
  __bf16* wob = (__bf16*)alloc((size_t)Cdim * Cdim * 2);
  float*  qf  = (float*)alloc(M * Cdim * 4);
  float*  kf  = (float*)alloc(M * KV * 4);
  float*  vf  = (float*)alloc(M * KV * 4);
  __bf16* Qb  = (__bf16*)alloc((size_t)B * N_HEADS_ * T * HEAD_DIM_ * 2);
  __bf16* Kb  = (__bf16*)alloc((size_t)B * N_KV_HEADS_ * T * HEAD_DIM_ * 2);
  __bf16* Vt  = (__bf16*)alloc((size_t)B * N_KV_HEADS_ * HEAD_DIM_ * T * 2);
  __bf16* Yb  = (__bf16*)alloc(M * Cdim * 2);

  // 1) downconvert inputs to bf16
  cvt_f32_bf16<<<1024, 256, 0, stream>>>(x,  xb,  M * Cdim);
  cvt_f32_bf16<<<1024, 256, 0, stream>>>(Wq, wqb, (size_t)Cdim * Cdim);
  cvt_f32_bf16<<<512,  256, 0, stream>>>(Wk, wkb, (size_t)KV * Cdim);
  cvt_f32_bf16<<<512,  256, 0, stream>>>(Wv, wvb, (size_t)KV * Cdim);
  cvt_f32_bf16<<<1024, 256, 0, stream>>>(Wo, wob, (size_t)Cdim * Cdim);

  // 2) q/k/v projections (WMMA bf16 -> f32)
  gemm_bf16<<<dim3(Cdim / 128, M / 128), 256, 0, stream>>>(xb, wqb, qf, (int)M, Cdim, Cdim);
  gemm_bf16<<<dim3(KV   / 128, M / 128), 256, 0, stream>>>(xb, wkb, kf, (int)M, KV,   Cdim);
  gemm_bf16<<<dim3(KV   / 128, M / 128), 256, 0, stream>>>(xb, wvb, vf, (int)M, KV,   Cdim);

  // 3) RMSNorm + RoPE; V transpose
  prep_kernel<<<dim3(T, N_HEADS_ + 2 * N_KV_HEADS_, B), 128, 0, stream>>>(
      qf, kf, vf, Qb, Kb, Vt, T);

  // 4) causal flash attention (WMMA)
  attn_kernel<<<dim3(T / 128, N_HEADS_, B), 128, 0, stream>>>(Qb, Kb, Vt, Yb, T);

  // 5) output projection into d_out (fp32)
  gemm_bf16<<<dim3(Cdim / 128, M / 128), 256, 0, stream>>>(
      Yb, wob, (float*)d_out, (int)M, Cdim, Cdim);
}
